// MambaTagger_62268435858075
// MI455X (gfx1250) — matrix operand order repaired and checked by reading, not compiled, not run on hardware
//
#include <hip/hip_runtime.h>
#include <hip/hip_bf16.h>

// ---------------------------------------------------------------------------
// MI455X (gfx1250, wave32) implementation.
// All large GEMMs -> v_wmma_f32_16x16x32_bf16, NT layout (both operands
// K-contiguous), 32x32 output tile per wave: 2 A-frags + 2 B-frags -> 4 WMMA
// per K-step (16 FLOP/byte per lane vs 8 for a 16x16 tile).
// fp32 accumulation + fp32 scalar math for LN / conv / softplus / SSM scan.
// ---------------------------------------------------------------------------

typedef __attribute__((ext_vector_type(16))) __bf16 v16bf;
typedef __attribute__((ext_vector_type(8)))  __bf16 v8bf;
typedef __attribute__((ext_vector_type(8)))  float  v8f;

#define B_   256
#define NLAB 20000
#define M_   3
#define D_   768
#define T_   5
#define K_   8
#define H_   256
#define DI_  512
#define S_   16
#define R_   16
#define NL_  4
#define KC_  4
#define ROWS (B_*T_*(K_+1))   /* 11520 token rows */
#define KP_  800              /* 769 padded to multiple of 32 */
#define DBL_LD 64             /* 48 (R+2S) padded to multiple of 32 */

// ---------------- block reduction helper (blockDim.x == 256) ---------------
__device__ __forceinline__ float block_reduce_sum(float v, float* red) {
    int tid = threadIdx.x;
    red[tid] = v; __syncthreads();
    for (int s = 128; s > 0; s >>= 1) {
        if (tid < s) red[tid] += red[tid + s];
        __syncthreads();
    }
    float total = red[0]; __syncthreads();
    return total;
}

// ---------------- softmax over modality logits (T x M, tiny) ---------------
__global__ void softmax_w_kernel(const float* __restrict__ mlog, float* __restrict__ w) {
    int t = threadIdx.x;
    if (t >= T_) return;
    float a = mlog[t*M_+0], b = mlog[t*M_+1], c = mlog[t*M_+2];
    float mx = fmaxf(a, fmaxf(b, c));
    float ea = __expf(a-mx), eb = __expf(b-mx), ec = __expf(c-mx);
    float s = ea + eb + ec;
    w[t*M_+0] = ea/s; w[t*M_+1] = eb/s; w[t*M_+2] = ec/s;
}

// ---------------- qn: per (t,b) row: project + L2-normalize -> bf16 --------
__global__ __launch_bounds__(256) void qn_kernel(const float* __restrict__ unl,
                                                 const float* __restrict__ w,
                                                 __bf16* __restrict__ qn) {
    __shared__ float red[256];
    int t = blockIdx.x / B_;
    int b = blockIdx.x % B_;
    float w0 = w[t*M_+0], w1 = w[t*M_+1], w2 = w[t*M_+2];
    const float* ub = unl + (long)b * M_ * D_;
    float vals[3]; float ss = 0.f;
    for (int j = 0; j < 3; ++j) {
        int d = threadIdx.x + j*256;
        float v = w0*ub[d] + w1*ub[D_+d] + w2*ub[2*D_+d];
        vals[j] = v; ss += v*v;
    }
    float total = block_reduce_sum(ss, red);
    float inv = rsqrtf(total + 1e-8f);
    __bf16* out = qn + ((long)t*B_ + b) * D_;
    for (int j = 0; j < 3; ++j) {
        int d = threadIdx.x + j*256;
        out[d] = (__bf16)(vals[j] * inv);
    }
}

// ---------------- kn: per n row: stage lab_e in LDS, all T projections -----
__global__ __launch_bounds__(256) void kn_kernel(const float* __restrict__ lab,
                                                 const float* __restrict__ w,
                                                 __bf16* __restrict__ kn) {
    __shared__ float le[M_*D_];
    __shared__ float red[256];
    int n = blockIdx.x;
    const float* src = lab + (long)n * M_ * D_;
    for (int i = threadIdx.x; i < M_*D_; i += 256) le[i] = src[i];
    __syncthreads();
    for (int t = 0; t < T_; ++t) {
        float w0 = w[t*M_+0], w1 = w[t*M_+1], w2 = w[t*M_+2];
        float vals[3]; float ss = 0.f;
        for (int j = 0; j < 3; ++j) {
            int d = threadIdx.x + j*256;
            float v = w0*le[d] + w1*le[D_+d] + w2*le[2*D_+d];
            vals[j] = v; ss += v*v;
        }
        float total = block_reduce_sum(ss, red);
        float inv = rsqrtf(total + 1e-8f);
        __bf16* out = kn + ((long)t*NLAB + n) * D_;
        for (int j = 0; j < 3; ++j) {
            int d = threadIdx.x + j*256;
            out[d] = (__bf16)(vals[j] * inv);
        }
        __syncthreads();
    }
}

// ---------------- generic NT bf16 WMMA GEMM, 32x32 tile per wave -----------
// C[M,N] (f32) = A[M,K](bf16, row-major) * Bt[N,K](bf16, row-major)^T
// optional bias[N], optional addsrc[M,N] (residual), optional relu.
// M%32==0, N%32==0, K%32==0. 4 waves per 128-thread block.
__global__ __launch_bounds__(128) void gemm_nt_bf16(
        float* __restrict__ C,
        const __bf16* __restrict__ A,
        const __bf16* __restrict__ Bt,
        int M, int N, int K,
        const float* __restrict__ bias,
        const float* __restrict__ addsrc,
        int relu) {
    int wave = threadIdx.x >> 5;
    int lane = threadIdx.x & 31;
    int ntiles = N >> 5;
    long tiles_total = (long)(M >> 5) * ntiles;
    long tile = (long)blockIdx.x * 4 + wave;
    if (tile >= tiles_total) return;            // whole-wave uniform exit
    int tn = (int)(tile % ntiles);
    int tm = (int)(tile / ntiles);
    int half = lane >> 4;
    int l16  = lane & 15;
    const __bf16* A0 = A  + (long)(tm*32 + l16) * K;        // m-rows
    const __bf16* A1 = A0 + 16L * K;
    const __bf16* B0 = Bt + (long)(tn*32 + l16) * K;        // n-rows
    const __bf16* B1 = B0 + 16L * K;
    int ka0 = half * 8;     // A frag: k in [ka0, ka0+8) and [16+ka0, 16+ka0+8)
    int kb0 = half * 8;     // B frag: k in [kb0, kb0+8) and [16+kb0, 16+kb0+8)
    v8f acc00 = {}, acc01 = {}, acc10 = {}, acc11 = {};
    for (int kk = 0; kk < K; kk += 32) {
        v8bf a0l = *(const v8bf*)(A0 + kk + ka0);
        v8bf a0h = *(const v8bf*)(A0 + kk + ka0 + 16);
        v8bf a1l = *(const v8bf*)(A1 + kk + ka0);
        v8bf a1h = *(const v8bf*)(A1 + kk + ka0 + 16);
        v8bf b0l = *(const v8bf*)(B0 + kk + kb0);
        v8bf b0h = *(const v8bf*)(B0 + kk + kb0 + 16);
        v8bf b1l = *(const v8bf*)(B1 + kk + kb0);
        v8bf b1h = *(const v8bf*)(B1 + kk + kb0 + 16);
        if (kk + 32 < K) {                       // emit global_prefetch_b8
            __builtin_prefetch(A0 + kk + 32 + ka0, 0, 3);
            __builtin_prefetch(A1 + kk + 32 + ka0, 0, 3);
            __builtin_prefetch(B0 + kk + 32 + kb0, 0, 3);
            __builtin_prefetch(B1 + kk + 32 + kb0, 0, 3);
        }
        v16bf a0, a1, b0, b1;
        #pragma unroll
        for (int i = 0; i < 8; ++i) {
            a0[i] = a0l[i]; a0[8+i] = a0h[i];
            a1[i] = a1l[i]; a1[8+i] = a1h[i];
            b0[i] = b0l[i]; b0[8+i] = b0h[i];
            b1[i] = b1l[i]; b1[8+i] = b1h[i];
        }
        acc00 = __builtin_amdgcn_wmma_f32_16x16x32_bf16(false, a0, false, b0, (short)0, acc00, false, false);
        acc01 = __builtin_amdgcn_wmma_f32_16x16x32_bf16(false, a0, false, b1, (short)0, acc01, false, false);
        acc10 = __builtin_amdgcn_wmma_f32_16x16x32_bf16(false, a1, false, b0, (short)0, acc10, false, false);
        acc11 = __builtin_amdgcn_wmma_f32_16x16x32_bf16(false, a1, false, b1, (short)0, acc11, false, false);
    }
    int col0 = tn*32 + l16;
    int col1 = col0 + 16;
    long row0 = (long)tm*32 + half*8;
    long row1 = row0 + 16;
    #pragma unroll
    for (int r = 0; r < 8; ++r) {
        float v00 = acc00[r], v01 = acc01[r], v10 = acc10[r], v11 = acc11[r];
        if (bias) {
            float bc0 = bias[col0], bc1 = bias[col1];
            v00 += bc0; v01 += bc1; v10 += bc0; v11 += bc1;
        }
        if (addsrc) {
            v00 += addsrc[(row0+r)*N + col0];
            v01 += addsrc[(row0+r)*N + col1];
            v10 += addsrc[(row1+r)*N + col0];
            v11 += addsrc[(row1+r)*N + col1];
        }
        if (relu) {
            v00 = fmaxf(v00, 0.f); v01 = fmaxf(v01, 0.f);
            v10 = fmaxf(v10, 0.f); v11 = fmaxf(v11, 0.f);
        }
        C[(row0+r)*N + col0] = v00;
        C[(row0+r)*N + col1] = v01;
        C[(row1+r)*N + col0] = v10;
        C[(row1+r)*N + col1] = v11;
    }
}

// ---------------- top-K(8) per sim row (one block per b, given t) ----------
__global__ __launch_bounds__(256) void topk_kernel(const float* __restrict__ sim,
                                                   int* __restrict__ outIdx,
                                                   int t) {
    __shared__ float sv[256];
    __shared__ int   si[256];
    __shared__ int   chosen[K_];
    int b = blockIdx.x;
    const float* row = sim + (long)b * NLAB;
    for (int k = 0; k < K_; ++k) {
        float best = -3.4e38f; int bi = 0x7fffffff;
        for (int i = threadIdx.x; i < NLAB; i += 256) {
            bool skip = false;
            for (int j = 0; j < k; ++j) if (chosen[j] == i) skip = true;
            if (skip) continue;
            float v = row[i];
            if (v > best || (v == best && i < bi)) { best = v; bi = i; }
        }
        sv[threadIdx.x] = best; si[threadIdx.x] = bi;
        __syncthreads();
        for (int s = 128; s > 0; s >>= 1) {
            if (threadIdx.x < s) {
                float v2 = sv[threadIdx.x + s]; int i2 = si[threadIdx.x + s];
                if (v2 > sv[threadIdx.x] ||
                    (v2 == sv[threadIdx.x] && i2 < si[threadIdx.x])) {
                    sv[threadIdx.x] = v2; si[threadIdx.x] = i2;
                }
            }
            __syncthreads();
        }
        if (threadIdx.x == 0) {
            chosen[k] = si[0];
            outIdx[((long)t*B_ + b)*K_ + k] = si[0];
        }
        __syncthreads();
    }
}

// ---------------- build padded bf16 token matrix [ROWS, KP_] ---------------
__global__ __launch_bounds__(256) void build_toks_kernel(
        const float* __restrict__ lab_e, const float* __restrict__ lab_t,
        const float* __restrict__ unl, const int* __restrict__ topk,
        __bf16* __restrict__ toks) {
    int row = blockIdx.x;            // (b*T + t)*9 + l
    int l  = row % (K_+1);
    int bt = row / (K_+1);
    int t  = bt % T_;
    int b  = bt / T_;
    __bf16* orow = toks + (long)row * KP_;
    const float third = 1.f/3.f;
    if (l < K_) {
        int idxv = topk[((long)t*B_ + b)*K_ + l];
        const float* e = lab_e + (long)idxv * M_ * D_;
        for (int d = threadIdx.x; d < D_; d += 256)
            orow[d] = (__bf16)((e[d] + e[D_+d] + e[2*D_+d]) * third);
        if (threadIdx.x == 0)
            orow[D_] = (__bf16)lab_t[(long)idxv*T_ + t];
    } else {
        const float* e = unl + (long)b * M_ * D_;
        for (int d = threadIdx.x; d < D_; d += 256)
            orow[d] = (__bf16)((e[d] + e[D_+d] + e[2*D_+d]) * third);
        if (threadIdx.x == 0)
            orow[D_] = (__bf16)0.f;
    }
    for (int d = D_ + 1 + threadIdx.x; d < KP_; d += 256)
        orow[d] = (__bf16)0.f;
}

// ------- transpose+convert weights: dst[Npad,Kpad] <- src[Ksrc,Nsrc] -------
__global__ void convT_kernel(__bf16* __restrict__ dst, const float* __restrict__ src,
                             int Ksrc, int Kpad, int Nsrc, int Npad) {
    long gid = (long)blockIdx.x * 256 + threadIdx.x;
    if (gid >= (long)Npad * Kpad) return;
    int n = (int)(gid / Kpad);
    int k = (int)(gid % Kpad);
    dst[gid] = (k < Ksrc && n < Nsrc) ? (__bf16)src[(long)k * Nsrc + n] : (__bf16)0.f;
}

// ---------------- LayerNorm (H=256 per row) -> f32 + bf16 ------------------
__global__ __launch_bounds__(256) void ln_kernel(const float* __restrict__ x,
                                                 const float* __restrict__ g,
                                                 const float* __restrict__ bb,
                                                 float* __restrict__ xn,
                                                 __bf16* __restrict__ xnb) {
    __shared__ float red[256];
    long r = blockIdx.x;
    int h = threadIdx.x;
    float v = x[r*H_ + h];
    float mean = block_reduce_sum(v, red) * (1.f/H_);
    float d = v - mean;
    float var = block_reduce_sum(d*d, red) * (1.f/H_);
    float o = d * rsqrtf(var + 1e-5f) * g[h] + bb[h];
    xn [r*H_ + h] = o;
    xnb[r*H_ + h] = (__bf16)o;
}

// ---------------- causal depthwise conv (KC=4) + SiLU ----------------------
__global__ __launch_bounds__(256) void conv_silu_kernel(
        const float* __restrict__ xz, const float* __restrict__ cw,
        const float* __restrict__ cb, float* __restrict__ xc,
        __bf16* __restrict__ xcb) {
    int n = blockIdx.x;   // 0..1279
    for (int rep = 0; rep < 2; ++rep) {
        int di = threadIdx.x + rep*256;
        float w0 = cw[di*KC_+0], w1 = cw[di*KC_+1], w2 = cw[di*KC_+2], w3 = cw[di*KC_+3];
        float bbv = cb[di];
        float p0 = 0.f, p1 = 0.f, p2 = 0.f;   // x[l-3], x[l-2], x[l-1]
        for (int l = 0; l < K_+1; ++l) {
            float cur = xz[((long)n*(K_+1) + l)*(2*DI_) + di];
            float v = bbv + p0*w0 + p1*w1 + p2*w2 + cur*w3;
            v = v / (1.f + __expf(-v));       // SiLU
            long o = ((long)n*(K_+1) + l)*DI_ + di;
            xc[o] = v; xcb[o] = (__bf16)v;
            p0 = p1; p1 = p2; p2 = cur;
        }
    }
}

// ---------------- dt = softplus(dt_raw @ Wdt + bdt)  (K=16, scalar) --------
__global__ void dt_kernel(const float* __restrict__ dbl, const float* __restrict__ Wdt,
                          const float* __restrict__ bdt, float* __restrict__ dt) {
    long gid = (long)blockIdx.x * 256 + threadIdx.x;
    if (gid >= (long)ROWS * DI_) return;
    long r = gid / DI_;
    int di = (int)(gid % DI_);
    const float* dr = dbl + r * DBL_LD;
    float acc = bdt[di];
    #pragma unroll
    for (int j = 0; j < R_; ++j) acc += dr[j] * Wdt[j*DI_ + di];
    dt[gid] = (acc > 20.f) ? acc : log1pf(__expf(acc));
}

// ---------------- selective-scan (L=9, S=16) + gate -> y (bf16) ------------
__global__ void scan_kernel(const float* __restrict__ dt, const float* __restrict__ xc,
                            const float* __restrict__ dbl, const float* __restrict__ xz,
                            const float* __restrict__ A_log, const float* __restrict__ Dpv,
                            __bf16* __restrict__ yb) {
    long gid = (long)blockIdx.x * 256 + threadIdx.x;
    if (gid >= (long)(B_*T_) * DI_) return;
    int n  = (int)(gid / DI_);
    int di = (int)(gid % DI_);
    float Av[S_];
    #pragma unroll
    for (int s = 0; s < S_; ++s) Av[s] = -__expf(A_log[(long)di*S_ + s]);
    float Dv = Dpv[di];
    float h[S_];
    #pragma unroll
    for (int s = 0; s < S_; ++s) h[s] = 0.f;
    for (int l = 0; l < K_+1; ++l) {
        long r = (long)n*(K_+1) + l;
        float dtv = dt[r*DI_ + di];
        float xcv = xc[r*DI_ + di];
        const float* drow = dbl + r*DBL_LD;
        float acc = 0.f;
        #pragma unroll
        for (int s = 0; s < S_; ++s) {
            float dA = __expf(dtv * Av[s]);
            h[s] = dA * h[s] + dtv * drow[R_ + s] * xcv;
            acc += h[s] * drow[R_ + S_ + s];
        }
        float z = xz[r*(2*DI_) + DI_ + di];
        float y = (acc + Dv*xcv) * (z / (1.f + __expf(-z)));
        yb[r*DI_ + di] = (__bf16)y;
    }
}

// ---------------- gather last token (l=8) rows -> bf16 ---------------------
__global__ void pick_ms_kernel(const float* __restrict__ x, __bf16* __restrict__ ms) {
    long gid = (long)blockIdx.x * 256 + threadIdx.x;
    if (gid >= (long)(B_*T_) * H_) return;
    long r = gid / H_;
    int h = (int)(gid % H_);
    ms[gid] = (__bf16)x[(r*(K_+1) + K_)*H_ + h];
}

// ---------------- final head: out = h1 @ W2 + b2 ---------------------------
__global__ void final_kernel(const float* __restrict__ h1, const float* __restrict__ W2,
                             const float* __restrict__ b2, float* __restrict__ out) {
    int r = blockIdx.x * 256 + threadIdx.x;
    if (r >= B_*T_) return;
    float acc = b2[0];
    #pragma unroll 4
    for (int hh = 0; hh < H_/2; ++hh) acc += h1[(long)r*(H_/2) + hh] * W2[hh];
    out[r] = acc;
}

// ===========================================================================
extern "C" void kernel_launch(void* const* d_in, const int* in_sizes, int n_in,
                              void* d_out, int out_size, void* d_ws, size_t ws_size,
                              hipStream_t stream) {
    const float* unl   = (const float*)d_in[0];
    const float* lab_e = (const float*)d_in[1];
    const float* lab_t = (const float*)d_in[2];
    const float* mlog  = (const float*)d_in[3];
    const float* Wp    = (const float*)d_in[4];
    const float* bp    = (const float*)d_in[5];
    const float* ln_g  = (const float*)d_in[6];
    const float* ln_b  = (const float*)d_in[7];
    const float* Win   = (const float*)d_in[8];
    const float* conv_w= (const float*)d_in[9];
    const float* conv_b= (const float*)d_in[10];
    const float* Wx    = (const float*)d_in[11];
    const float* Wdt   = (const float*)d_in[12];
    const float* bdt   = (const float*)d_in[13];
    const float* A_log = (const float*)d_in[14];
    const float* Dp    = (const float*)d_in[15];
    const float* Wout  = (const float*)d_in[16];
    const float* W1    = (const float*)d_in[17];
    const float* b1    = (const float*)d_in[18];
    const float* W2    = (const float*)d_in[19];
    const float* b2    = (const float*)d_in[20];
    float* out = (float*)d_out;

    // ---- workspace bump allocator (~350 MB total) ----
    size_t off = 0;
    auto alloc = [&](size_t bytes) -> char* {
        char* p = (char*)d_ws + off;
        off += (bytes + 255) & ~(size_t)255;
        return p;
    };
    float*  w_sm   = (float*) alloc(T_*M_*4);
    __bf16* qn     = (__bf16*)alloc((size_t)T_*B_*D_*2);
    __bf16* kn     = (__bf16*)alloc((size_t)T_*NLAB*D_*2);
    float*  sim    = (float*) alloc((size_t)B_*NLAB*4);        // reused per t
    int*    tki    = (int*)   alloc((size_t)T_*B_*K_*4);
    __bf16* toks   = (__bf16*)alloc((size_t)ROWS*KP_*2);
    __bf16* WpT    = (__bf16*)alloc((size_t)H_*KP_*2);
    __bf16* WinT   = (__bf16*)alloc((size_t)NL_*2*DI_*H_*2);
    __bf16* WxT    = (__bf16*)alloc((size_t)NL_*DBL_LD*DI_*2); // N padded 48->64
    __bf16* WoutT  = (__bf16*)alloc((size_t)NL_*H_*DI_*2);
    __bf16* W1T    = (__bf16*)alloc((size_t)(H_/2)*H_*2);
    float*  x      = (float*) alloc((size_t)ROWS*H_*4);
    float*  xn     = (float*) alloc((size_t)ROWS*H_*4);
    __bf16* xnb    = (__bf16*)alloc((size_t)ROWS*H_*2);
    float*  xz     = (float*) alloc((size_t)ROWS*2*DI_*4);
    float*  xc     = (float*) alloc((size_t)ROWS*DI_*4);
    __bf16* xcb    = (__bf16*)alloc((size_t)ROWS*DI_*2);
    float*  dbl    = (float*) alloc((size_t)ROWS*DBL_LD*4);    // row stride 64
    float*  dtb    = (float*) alloc((size_t)ROWS*DI_*4);
    __bf16* yb     = (__bf16*)alloc((size_t)ROWS*DI_*2);
    __bf16* ms     = (__bf16*)alloc((size_t)B_*T_*H_*2);
    float*  h1     = (float*) alloc((size_t)B_*T_*(H_/2)*4);
    (void)in_sizes; (void)n_in; (void)out_size; (void)ws_size;

    auto gemm = [&](float* C, const __bf16* A, const __bf16* Bt, int M, int N, int K,
                    const float* bias, const float* addsrc, int relu) {
        long tiles = (long)(M/32) * (N/32);
        int blocks = (int)((tiles + 3) / 4);
        gemm_nt_bf16<<<blocks, 128, 0, stream>>>(C, A, Bt, M, N, K, bias, addsrc, relu);
    };

    // 1) modality softmax, normalized query/key tensors (bf16)
    softmax_w_kernel<<<1, 32, 0, stream>>>(mlog, w_sm);
    qn_kernel<<<T_*B_, 256, 0, stream>>>(unl, w_sm, qn);
    kn_kernel<<<NLAB, 256, 0, stream>>>(lab_e, w_sm, kn);

    // 2) weight transpose+convert (NT layout, K-contiguous, K/N zero-padded)
    convT_kernel<<<(H_*KP_ + 255)/256, 256, 0, stream>>>(WpT, Wp, D_+1, KP_, H_, H_);
    for (int i = 0; i < NL_; ++i) {
        convT_kernel<<<(2*DI_*H_ + 255)/256, 256, 0, stream>>>(
            WinT + (size_t)i*2*DI_*H_, Win + (size_t)i*H_*2*DI_, H_, H_, 2*DI_, 2*DI_);
        convT_kernel<<<(DBL_LD*DI_ + 255)/256, 256, 0, stream>>>(
            WxT + (size_t)i*DBL_LD*DI_, Wx + (size_t)i*DI_*(R_+2*S_), DI_, DI_, R_+2*S_, DBL_LD);
        convT_kernel<<<(H_*DI_ + 255)/256, 256, 0, stream>>>(
            WoutT + (size_t)i*H_*DI_, Wout + (size_t)i*DI_*H_, DI_, DI_, H_, H_);
    }
    convT_kernel<<<((H_/2)*H_ + 255)/256, 256, 0, stream>>>(W1T, W1, H_, H_, H_/2, H_/2);

    // 3) per-trait similarity GEMM (WMMA, 39 GF total) + top-K
    for (int t = 0; t < T_; ++t) {
        gemm(sim, qn + (size_t)t*B_*D_, kn + (size_t)t*NLAB*D_,
             B_, NLAB, D_, nullptr, nullptr, 0);
        topk_kernel<<<B_, 256, 0, stream>>>(sim, tki, t);
    }

    // 4) tokens -> input projection (WMMA, K padded 769->800)
    build_toks_kernel<<<ROWS, 256, 0, stream>>>(lab_e, lab_t, unl, tki, toks);
    gemm(x, toks, WpT, ROWS, H_, KP_, bp, nullptr, 0);

    // 5) four Mamba layers
    for (int i = 0; i < NL_; ++i) {
        ln_kernel<<<ROWS, 256, 0, stream>>>(x, ln_g + (size_t)i*H_, ln_b + (size_t)i*H_, xn, xnb);
        gemm(xz, xnb, WinT + (size_t)i*2*DI_*H_, ROWS, 2*DI_, H_, nullptr, nullptr, 0);
        conv_silu_kernel<<<B_*T_, 256, 0, stream>>>(
            xz, conv_w + (size_t)i*DI_*KC_, conv_b + (size_t)i*DI_, xc, xcb);
        gemm(dbl, xcb, WxT + (size_t)i*DBL_LD*DI_, ROWS, DBL_LD, DI_, nullptr, nullptr, 0);
        dt_kernel<<<((long)ROWS*DI_ + 255)/256, 256, 0, stream>>>(
            dbl, Wdt + (size_t)i*R_*DI_, bdt + (size_t)i*DI_, dtb);
        scan_kernel<<<((long)B_*T_*DI_ + 255)/256, 256, 0, stream>>>(
            dtb, xc, dbl, xz, A_log + (size_t)i*DI_*S_, Dp + (size_t)i*DI_, yb);
        // x_new = xn + y @ Wout  (residual is the *normalized* input, per ref)
        gemm(x, yb, WoutT + (size_t)i*H_*DI_, ROWS, H_, DI_, nullptr, xn, 0);
    }

    // 6) head MLP
    pick_ms_kernel<<<((long)B_*T_*H_ + 255)/256, 256, 0, stream>>>(x, ms);
    gemm(h1, ms, W1T, B_*T_, H_/2, H_, b1, nullptr, 1);
    final_kernel<<<(B_*T_ + 255)/256, 256, 0, stream>>>(h1, W2, b2, out);
}
